// Attention_68642167325252
// MI455X (gfx1250) — matrix operand order repaired and checked by reading, not compiled, not run on hardware
//
#include <hip/hip_runtime.h>
#include <hip/hip_bf16.h>
#include <cstdint>

// ---------------------------------------------------------------------------
// Types for CDNA5 WMMA (wave32): bf16 A/B fragments (16 x bf16 = 8 VGPRs),
// f32 C/D accumulator (8 VGPRs).
// ---------------------------------------------------------------------------
typedef __bf16 v16bf __attribute__((ext_vector_type(16)));
typedef float  v8f   __attribute__((ext_vector_type(8)));
typedef unsigned int v4u __attribute__((ext_vector_type(4)));
typedef unsigned int v8u __attribute__((ext_vector_type(8)));

union FragBF {
    uint4 q[2];   // 32 bytes
    v16bf v;
};

__device__ __forceinline__ unsigned short f32_to_bf16(float f) {
    union { float f; uint32_t u; } x{f};
    uint32_t u = x.u;
    uint32_t r = u + 0x7FFFu + ((u >> 16) & 1u);   // round-to-nearest-even
    return (unsigned short)(r >> 16);
}

// ---------------------------------------------------------------------------
// CDNA5 async global -> LDS copy (16B per lane, tracked by ASYNCcnt).
// ---------------------------------------------------------------------------
__device__ __forceinline__ void async_copy_b128(unsigned lds_off, const void* g) {
    asm volatile("global_load_async_to_lds_b128 %0, %1, off"
                 :: "v"(lds_off), "v"((unsigned long long)(uintptr_t)g)
                 : "memory");
}
__device__ __forceinline__ void wait_async0() {
    asm volatile("s_wait_asynccnt 0x0" ::: "memory");
}
__device__ __forceinline__ unsigned lds_off_of(const void* p) {
    return (unsigned)(uintptr_t)p;   // low 32 bits = LDS byte offset
}

// ---------------------------------------------------------------------------
// CDNA5 Tensor Data Mover: one-instruction 2D tile DMA global -> LDS.
// Builds a 2D D# (groups 0/1, VADDR2/3 = NULL) per CDNA5 ISA:
//  - data_size = 1 (2-byte bf16)
//  - pad_enable: insert pad_amount DWORDs every pad_interval DWORDs so the
//    LDS destination keeps our +8-element row padding (row stride tile_w+8).
//  - tensor_dim0_stride = stride_elems (row pitch of source matrix).
// Issued per-wave (EXEC ignored); tracked by TENSORcnt.
// ---------------------------------------------------------------------------
__device__ __forceinline__ void tdm_load_2d(unsigned lds_off, const void* g,
                                            unsigned tile_w, unsigned tile_h,
                                            unsigned stride_elems) {
    unsigned long long ga = (unsigned long long)(uintptr_t)g;
    v4u g0;
    g0[0] = 1u;                                   // count=1, user descriptor
    g0[1] = lds_off;                              // LDS byte address
    g0[2] = (unsigned)ga;                         // global_addr[31:0]
    g0[3] = ((unsigned)(ga >> 32) & 0x01FFFFFFu)  // global_addr[56:32]
            | (2u << 30);                         // type = 2 ("image")
    const unsigned padInt = (31u - (unsigned)__builtin_clz(tile_w)) - 2u; // log2(w)-2: row DWORDs
    const unsigned dim1   = 1u << 20;             // tensor_dim1 (no OOB)
    v8u g1;
    g1[0] = (1u << 16)            // data_size = 1 (2 bytes)
          | (1u << 20)            // pad_enable
          | (padInt << 22)        // pad_interval: tile_w/2 DWORDs per row
          | (3u << 25);           // pad_amount: 4 DWORDs (= 8 bf16 elems)
    g1[1] = (stride_elems & 0xFFFFu) << 16;                    // tensor_dim0[15:0]
    g1[2] = ((stride_elems >> 16) & 0xFFFFu)                   // tensor_dim0[31:16]
          | ((dim1 & 0xFFFFu) << 16);                          // tensor_dim1[15:0]
    g1[3] = ((dim1 >> 16) & 0xFFFFu)                           // tensor_dim1[31:16]
          | (tile_w << 16);                                    // tile_dim0
    g1[4] = tile_h & 0xFFFFu;                                  // tile_dim1 (tile_dim2=0)
    g1[5] = stride_elems;                                      // tensor_dim0_stride[31:0]
    g1[6] = 0u;                                                // stride[47:32], d1stride[15:0]
    g1[7] = 0u;                                                // d1stride[47:16]
    asm volatile("tensor_load_to_lds %0, %1" :: "s"(g0), "s"(g1) : "memory");
}
__device__ __forceinline__ void wait_tensor0() {
    asm volatile("s_wait_tensorcnt 0x0" ::: "memory");
}

#define ATT_SCALE 0.125f   // HD=64 -> 1/sqrt(64)

// ---------------------------------------------------------------------------
// fp32 -> bf16 conversion (grid-stride)
// ---------------------------------------------------------------------------
__global__ void convert_f32_to_bf16(const float* __restrict__ src,
                                    unsigned short* __restrict__ dst, int n) {
    int i = blockIdx.x * blockDim.x + threadIdx.x;
    int stride = gridDim.x * blockDim.x;
    for (; i < n; i += stride) dst[i] = f32_to_bf16(src[i]);
}

// ---------------------------------------------------------------------------
// GEMM: Y[M,N] = A[M,K] (bf16, row-major) * W[N,K]^T (bf16, row-major)
// Block: 128x128 tile, 256 threads (8 waves). Wave: 32x64 = 2x4 WMMA tiles.
// BK = 32. Double-buffered LDS staged by the Tensor Data Mover: wave 0 issues
// one tensor_load_to_lds per matrix per K-step; the DMA of tile t+1 overlaps
// the WMMAs of tile t. One barrier per K-step.
// ---------------------------------------------------------------------------
template <bool OUT_BF16>
__global__ __launch_bounds__(256) void gemm_bf16_wmma(
    const unsigned short* __restrict__ A,
    const unsigned short* __restrict__ W,
    const float* __restrict__ bias,
    unsigned short* __restrict__ outB,
    float* __restrict__ outF,
    int M, int N, int K)
{
    constexpr int BM = 128, BN = 128, BK = 32, LDT = BK + 8; // row stride 80B (16B aligned)
    __shared__ unsigned short As[2][BM * LDT];
    __shared__ unsigned short Bs[2][BN * LDT];

    const int t    = threadIdx.x;
    const int lane = t & 31;
    const int w    = t >> 5;
    const int l16  = lane & 15;
    const int hi   = lane >> 4;          // 0: lanes 0-15, 1: lanes 16-31

    const int rowBase = blockIdx.y * BM;
    const int colBase = blockIdx.x * BN;
    const int wr = (w >> 1) * 32;        // wave row offset within block tile
    const int wc = (w & 1) * 64;         // wave col offset within block tile

    v8f acc[2][4];
    #pragma unroll
    for (int mi = 0; mi < 2; ++mi)
        #pragma unroll
        for (int ni = 0; ni < 4; ++ni)
            #pragma unroll
            for (int e = 0; e < 8; ++e) acc[mi][ni][e] = 0.0f;

    // One TDM descriptor per matrix tile (wave 0 only; per-wave DMA).
    auto stage = [&](int buf, int k0) {
        if (w == 0) {
            tdm_load_2d(lds_off_of(&As[buf][0]),
                        &A[(size_t)rowBase * K + k0], BK, BM, (unsigned)K);
            tdm_load_2d(lds_off_of(&Bs[buf][0]),
                        &W[(size_t)colBase * K + k0], BK, BN, (unsigned)K);
        }
    };

    const int nk = K / BK;
    stage(0, 0);

    for (int kt = 0; kt < nk; ++kt) {
        const int buf = kt & 1;
        if (w == 0) wait_tensor0();   // wave 0's DMAs for `buf` landed in LDS
        __syncthreads();              // visible to all; buf^1 reads finished
        if (kt + 1 < nk) stage(buf ^ 1, (kt + 1) * BK);  // overlaps WMMAs below

        FragBF a[2], b[4];
        #pragma unroll
        for (int mi = 0; mi < 2; ++mi) {
            const unsigned short* p = &As[buf][(wr + mi * 16 + l16) * LDT];
            a[mi].q[0] = *(const uint4*)&p[hi * 16];
            a[mi].q[1] = *(const uint4*)&p[hi * 16 + 8];
        }
        #pragma unroll
        for (int ni = 0; ni < 4; ++ni) {
            const unsigned short* p = &Bs[buf][(wc + ni * 16 + l16) * LDT + hi * 16];
            b[ni].q[0] = *(const uint4*)&p[0];
            b[ni].q[1] = *(const uint4*)&p[8];
        }
        #pragma unroll
        for (int mi = 0; mi < 2; ++mi)
            #pragma unroll
            for (int ni = 0; ni < 4; ++ni)
                acc[mi][ni] = __builtin_amdgcn_wmma_f32_16x16x32_bf16(
                    false, a[mi].v, false, b[ni].v, (short)0, acc[mi][ni],
                    false, false);
    }

    // Store: C/D layout -> VGPR j: lanes 0-15 row j, lanes 16-31 row j+8; col = l16.
    #pragma unroll
    for (int mi = 0; mi < 2; ++mi) {
        #pragma unroll
        for (int ni = 0; ni < 4; ++ni) {
            int col = colBase + wc + ni * 16 + l16;
            float bv = 0.0f;
            if (!OUT_BF16 && bias) bv = bias[col];
            #pragma unroll
            for (int j = 0; j < 8; ++j) {
                int row = rowBase + wr + mi * 16 + j + hi * 8;
                float vv = acc[mi][ni][j] + bv;
                if (OUT_BF16) outB[(size_t)row * N + col] = f32_to_bf16(vv);
                else          outF[(size_t)row * N + col] = vv;
            }
        }
    }
}

// ---------------------------------------------------------------------------
// Flash attention: one block per (b, h, 128-query tile). 8 waves; wave owns
// 16 query rows. Online softmax over key chunks of 64.
//  - Q tile staged with per-lane global_load_async_to_lds_b128 (ASYNCcnt)
//  - K chunks staged by the Tensor Data Mover (TENSORcnt, wave 0)
//  - V transposed into LDS for contiguous B-fragments
// QKV layout: [B*N, 3*C] bf16. Output O: [B*N, C] bf16.
// ---------------------------------------------------------------------------
__global__ __launch_bounds__(256) void attn_wmma(
    const unsigned short* __restrict__ QKV,
    unsigned short* __restrict__ O)
{
    constexpr int HD = 64, NKEY = 1024, CK = 64;
    constexpr int LQ = HD + 8;   // Q/K row stride (elems) -> 144B, 16B aligned
    constexpr int LP = CK + 8;   // Vt/P row stride (elems) -> 144B, 16B aligned
    constexpr int C3 = 3 * 1024;

    __shared__ unsigned short Qs[128 * LQ];     // 18 KB
    __shared__ unsigned short Ks[CK * LQ];      //  9 KB
    __shared__ unsigned short VTs[HD * LP];     //  9 KB (V transposed: [d][key])
    __shared__ unsigned short Ps[8 * 16 * LP];  // 18 KB (per-wave P tiles)

    const int qt = blockIdx.x;   // query tile 0..7
    const int h  = blockIdx.y;   // head 0..15
    const int b  = blockIdx.z;   // batch 0..7

    const int t    = threadIdx.x;
    const int lane = t & 31;
    const int w    = t >> 5;
    const int l16  = lane & 15;
    const int hi   = lane >> 4;

    const size_t rowQ0 = (size_t)b * 1024 + (size_t)qt * 128;
    const size_t rowK0 = (size_t)b * 1024;
    const int colQ = h * HD, colK = 1024 + h * HD, colV = 2048 + h * HD;

    // Async-stage Q tile (128 x 64 bf16): 1024 16B-chunks.
    {
        const unsigned qBase = lds_off_of(&Qs[0]);
        for (int chunk = t; chunk < 128 * 8; chunk += 256) {
            int r = chunk >> 3, seg = (chunk & 7) * 8;
            async_copy_b128(qBase + (unsigned)((r * LQ + seg) * 2),
                            &QKV[(rowQ0 + r) * C3 + colQ + seg]);
        }
    }

    float mrow[8], lrow[8];
    v8f o[4];
    #pragma unroll
    for (int j = 0; j < 8; ++j) { mrow[j] = -1e30f; lrow[j] = 0.0f; }
    #pragma unroll
    for (int dn = 0; dn < 4; ++dn)
        #pragma unroll
        for (int e = 0; e < 8; ++e) o[dn][e] = 0.0f;

    const int qr = w * 16;  // this wave's query-row base within the tile

    for (int key0 = 0; key0 < NKEY; key0 += CK) {
        __syncthreads();     // previous chunk fully consumed before overwrite
        // K chunk (64 x 64 bf16): single TDM tile DMA (wave 0).
        if (w == 0)
            tdm_load_2d(lds_off_of(&Ks[0]),
                        &QKV[(rowK0 + key0) * C3 + colK], HD, CK, C3);
        // V chunk, stored transposed VTs[d][key] so B-fragments are contiguous.
        for (int idx = t; idx < CK * HD; idx += 256) {
            int r = idx >> 6;     // key within chunk
            int d = idx & 63;
            VTs[d * LP + r] = QKV[(rowK0 + key0 + r) * C3 + colV + d];
        }
        wait_async0();                 // Q asyncs landed (first chunk)
        if (w == 0) wait_tensor0();    // K DMA landed
        __syncthreads();               // all staging visible to all waves

        // ----- S = Q * K^T (16 x 64 per wave), 4 col tiles x 2 K-steps -----
        v8f s[4];
        #pragma unroll
        for (int tn = 0; tn < 4; ++tn)
            #pragma unroll
            for (int e = 0; e < 8; ++e) s[tn][e] = 0.0f;

        #pragma unroll
        for (int ks = 0; ks < 2; ++ks) {
            FragBF aq;
            const unsigned short* pq = &Qs[(qr + l16) * LQ + ks * 32];
            aq.q[0] = *(const uint4*)&pq[hi * 16];
            aq.q[1] = *(const uint4*)&pq[hi * 16 + 8];
            #pragma unroll
            for (int tn = 0; tn < 4; ++tn) {
                FragBF bk;
                const unsigned short* pk =
                    &Ks[(tn * 16 + l16) * LQ + ks * 32 + hi * 16];
                bk.q[0] = *(const uint4*)&pk[0];
                bk.q[1] = *(const uint4*)&pk[8];
                s[tn] = __builtin_amdgcn_wmma_f32_16x16x32_bf16(
                    false, aq.v, false, bk.v, (short)0, s[tn], false, false);
            }
        }

        // ----- online softmax (per-row stats; row = qr + j + hi*8) -----
        float alpha[8];
        #pragma unroll
        for (int j = 0; j < 8; ++j) {
            float mx = -1e30f;
            #pragma unroll
            for (int tn = 0; tn < 4; ++tn) mx = fmaxf(mx, s[tn][j]);
            mx *= ATT_SCALE;
            #pragma unroll
            for (int off = 1; off < 16; off <<= 1)
                mx = fmaxf(mx, __shfl_xor(mx, off, 16));
            float mnew = fmaxf(mrow[j], mx);
            alpha[j] = __expf(mrow[j] - mnew);
            mrow[j] = mnew;
        }
        #pragma unroll
        for (int j = 0; j < 8; ++j) {
            float rs = 0.0f;
            #pragma unroll
            for (int tn = 0; tn < 4; ++tn) {
                float p = __expf(s[tn][j] * ATT_SCALE - mrow[j]);
                rs += p;
                // P in A-friendly row-major layout (per-wave region)
                Ps[(w * 16 + j + hi * 8) * LP + tn * 16 + l16] = f32_to_bf16(p);
            }
            #pragma unroll
            for (int off = 1; off < 16; off <<= 1)
                rs += __shfl_xor(rs, off, 16);
            lrow[j] = lrow[j] * alpha[j] + rs;
            #pragma unroll
            for (int dn = 0; dn < 4; ++dn) o[dn][j] *= alpha[j];
        }

        // ----- O += P * V (K-dim = 64 keys, 2 K-steps; same-wave LDS, in-order)
        #pragma unroll
        for (int s2 = 0; s2 < 2; ++s2) {
            FragBF ap;
            const unsigned short* pp = &Ps[(w * 16 + l16) * LP + s2 * 32];
            ap.q[0] = *(const uint4*)&pp[hi * 16];
            ap.q[1] = *(const uint4*)&pp[hi * 16 + 8];
            #pragma unroll
            for (int dn = 0; dn < 4; ++dn) {
                FragBF bv;
                const unsigned short* pv =
                    &VTs[(dn * 16 + l16) * LP + s2 * 32 + hi * 16];
                bv.q[0] = *(const uint4*)&pv[0];
                bv.q[1] = *(const uint4*)&pv[8];
                o[dn] = __builtin_amdgcn_wmma_f32_16x16x32_bf16(
                    false, ap.v, false, bv.v, (short)0, o[dn], false, false);
            }
        }
    }

    // ----- normalize and write O (bf16, [B*N, C]) -----
    #pragma unroll
    for (int dn = 0; dn < 4; ++dn) {
        int col = h * HD + dn * 16 + l16;
        #pragma unroll
        for (int j = 0; j < 8; ++j) {
            size_t row = rowQ0 + qr + j + hi * 8;
            O[row * 1024 + col] = f32_to_bf16(o[dn][j] / lrow[j]);
        }
    }
}

// ---------------------------------------------------------------------------
// Host launcher
// ---------------------------------------------------------------------------
extern "C" void kernel_launch(void* const* d_in, const int* in_sizes, int n_in,
                              void* d_out, int out_size, void* d_ws, size_t ws_size,
                              hipStream_t stream) {
    (void)in_sizes; (void)n_in; (void)out_size; (void)ws_size;
    const float* x      = (const float*)d_in[0];   // [8,1024,1024]
    const float* w_qkv  = (const float*)d_in[1];   // [3072,1024]
    const float* w_proj = (const float*)d_in[2];   // [1024,1024]
    const float* b_proj = (const float*)d_in[3];   // [1024]
    float* out = (float*)d_out;                    // [8,1024,1024] f32

    const int Bv = 8, Nv = 1024, C = 1024;
    const int M = Bv * Nv;                         // 8192

    // Workspace layout (bf16 = unsigned short), ~92 MB total:
    unsigned short* Xb     = (unsigned short*)d_ws;             // M*C
    unsigned short* Wqkvb  = Xb     + (size_t)M * C;            // 3C*C
    unsigned short* Wprojb = Wqkvb  + (size_t)3 * C * C;        // C*C
    unsigned short* QKVb   = Wprojb + (size_t)C * C;            // M*3C
    unsigned short* Ob     = QKVb   + (size_t)M * 3 * C;        // M*C

    convert_f32_to_bf16<<<512, 256, 0, stream>>>(x,      Xb,     M * C);
    convert_f32_to_bf16<<<512, 256, 0, stream>>>(w_qkv,  Wqkvb,  3 * C * C);
    convert_f32_to_bf16<<<256, 256, 0, stream>>>(w_proj, Wprojb, C * C);

    // QKV projection: [8192,3072] = Xb * Wqkv^T  (bf16 out)
    dim3 g1(3 * C / 128, M / 128);
    gemm_bf16_wmma<true><<<g1, 256, 0, stream>>>(
        Xb, Wqkvb, nullptr, QKVb, nullptr, M, 3 * C, C);

    // Attention: grid (qtile=8, head=16, batch=8)
    dim3 g2(Nv / 128, 16, Bv);
    attn_wmma<<<g2, 256, 0, stream>>>(QKVb, Ob);

    // Output projection + bias: f32 out
    dim3 g3(C / 128, M / 128);
    gemm_bf16_wmma<false><<<g3, 256, 0, stream>>>(
        Ob, Wprojb, b_proj, nullptr, out, M, C, C);
}
